// Ricci_11596411700024
// MI455X (gfx1250) — matrix operand order repaired and checked by reading, not compiled, not run on hardware
//
#include <hip/hip_runtime.h>
#include <hip/hip_bf16.h>
#include <stdint.h>

// Ricci edge-network, MI455X (gfx1250) wave32 WMMA implementation.
// B=8, N=128, D=256 hardcoded per reference setup_inputs().

#define ALPHA 0.2f
#define BETA  0.5f
#define BIG   1e9f

typedef __attribute__((ext_vector_type(16))) _Float16 v16h;
typedef __attribute__((ext_vector_type(8)))  float    v8f;
typedef __attribute__((ext_vector_type(4)))  float    v4f;   // native vec for NT builtins

union Frag16 { uint4 q[2]; v16h h; };

__device__ __forceinline__ float gelu_exact(float x) {
  return 0.5f * x * (1.0f + erff(x * 0.70710678118654752f));
}

__device__ __forceinline__ uint32_t pack2h(float a, float b) {
  union { _Float16 h[2]; uint32_t u; } t;
  t.h[0] = (_Float16)a; t.h[1] = (_Float16)b;
  return t.u;
}

__device__ __forceinline__ void ld8(const float* __restrict__ p, int lane, float* o) {
  float4 a = ((const float4*)p)[lane * 2];
  float4 b = ((const float4*)p)[lane * 2 + 1];
  o[0]=a.x; o[1]=a.y; o[2]=a.z; o[3]=a.w; o[4]=b.x; o[5]=b.y; o[6]=b.z; o[7]=b.w;
}

// Non-temporal (TH=NT) variants for touch-once streaming data: keep the
// 192MB L2 free for the reused P/Q/top3 working set.
__device__ __forceinline__ void ld8_nt(const float* __restrict__ p, int lane, float* o) {
  v4f a = __builtin_nontemporal_load(&((const v4f*)p)[lane * 2]);
  v4f b = __builtin_nontemporal_load(&((const v4f*)p)[lane * 2 + 1]);
  o[0]=a.x; o[1]=a.y; o[2]=a.z; o[3]=a.w; o[4]=b.x; o[5]=b.y; o[6]=b.z; o[7]=b.w;
}

__device__ __forceinline__ void st8_nt(float* __restrict__ p, int lane, const float* v) {
  v4f a, b;
  a.x=v[0]; a.y=v[1]; a.z=v[2]; a.w=v[3]; b.x=v[4]; b.y=v[5]; b.z=v[6]; b.w=v[7];
  __builtin_nontemporal_store(a, &((v4f*)p)[lane * 2]);
  __builtin_nontemporal_store(b, &((v4f*)p)[lane * 2 + 1]);
}

// ---------------------------------------------------------------------------
// K0: pre-swizzle weights into WMMA B-fragment layout (f16 pairs).
// B-frag (16-bit, 32x16): lane half selects K+8; VGPR v holds K pair 2v (v<4) / 16+2(v-4) (v>=4);
// N = lane&15. Storage: [kt][nt][lane][8 dwords] -> contiguous 32B per lane.
// w1f : W_e2v1 [256,128]    -> kt:8, nt:8   (16384 dwords, 64 KB)
// wcf : [Wtop|Wbot] [256,512]-> kt:8, nt:32 (65536 dwords, 256 KB)
// ---------------------------------------------------------------------------
__global__ void k_swizzle(const float* __restrict__ Wn2e,   // [512,256]
                          const float* __restrict__ W1,     // [256,128]
                          uint32_t* __restrict__ w1f,
                          uint32_t* __restrict__ wcf) {
  int t = blockIdx.x * blockDim.x + threadIdx.x;
  if (t < 16384) {
    int v = t & 7, lane = (t >> 3) & 31, nt = (t >> 8) & 7, kt = t >> 11;
    int K = kt * 32 + ((v & 4) << 2) + 2 * (v & 3) + ((lane >> 4) << 3);
    int N = nt * 16 + (lane & 15);
    w1f[t] = pack2h(W1[K * 128 + N], W1[(K + 1) * 128 + N]);
  } else if (t < 16384 + 65536) {
    int u = t - 16384;
    int v = u & 7, lane = (u >> 3) & 31, nt = (u >> 8) & 31, kt = u >> 13;
    int K = kt * 32 + ((v & 4) << 2) + 2 * (v & 3) + ((lane >> 4) << 3);
    int n = nt * 16 + (lane & 15);
    float a, b;
    if (n < 256) { a = Wn2e[K * 256 + n];                 b = Wn2e[(K + 1) * 256 + n]; }
    else         { a = Wn2e[(K + 256) * 256 + (n - 256)]; b = Wn2e[(K + 257) * 256 + (n - 256)]; }
    wcf[u] = pack2h(a, b);
  }
}

// ---------------------------------------------------------------------------
// K1: [P|Q] = X @ [Wtop|Wbot]   (X: [1024,256] fp32 -> f16 A tiles in LDS)
// One block (4 waves) per 16-row M tile; each wave covers 8 N-tiles.
// ---------------------------------------------------------------------------
__global__ void __launch_bounds__(128) k_pq(const float* __restrict__ X,
                                            const uint32_t* __restrict__ wcf,
                                            float* __restrict__ P,
                                            float* __restrict__ Q) {
  __shared__ __align__(16) uint16_t xt[16 * 256];
  int tid = threadIdx.x;
  int base = blockIdx.x * 16;
  for (int dw = tid; dw < 2048; dw += 128) {
    int row = dw >> 7, cp = (dw & 127) * 2;
    ((uint32_t*)xt)[dw] = pack2h(X[(base + row) * 256 + cp],
                                 X[(base + row) * 256 + cp + 1]);
  }
  __syncthreads();
  int lane = tid & 31, wave = tid >> 5;
  int half = lane >> 4, M = lane & 15;
  // A fragments: per documented 16-bit A 16x32 layout, lane half h reads
  // K = kt*32 + 8h .. +7 and K = kt*32 + 16 + 8h .. +7 (two b128s).
  Frag16 afrag[8];
  const uint4* lx = (const uint4*)xt;
#pragma unroll
  for (int kt = 0; kt < 8; ++kt) {
    int el = M * 256 + kt * 32 + half * 8;
    afrag[kt].q[0] = lx[el >> 3];
    afrag[kt].q[1] = lx[(el + 16) >> 3];
  }
  const uint4* wf = (const uint4*)wcf;
  for (int s = 0; s < 8; ++s) {
    int nt = wave * 8 + s;
    v8f acc = {0.f, 0.f, 0.f, 0.f, 0.f, 0.f, 0.f, 0.f};
#pragma unroll
    for (int kt = 0; kt < 8; ++kt) {
      Frag16 bf;
      int fi = ((kt * 32 + nt) * 32 + lane) * 2;
      bf.q[0] = wf[fi];
      bf.q[1] = wf[fi + 1];
      acc = __builtin_amdgcn_wmma_f32_16x16x32_f16(false, afrag[kt].h,
                                                   false, bf.h,
                                                   (short)0, acc, false, false);
    }
    int ncol = nt * 16 + M;
#pragma unroll
    for (int r = 0; r < 8; ++r) {
      int row = base + r + half * 8;           // C layout: lanes>=16 hold M=r+8
      if (ncol < 256) P[row * 256 + ncol] = acc[r];
      else            Q[row * 256 + (ncol - 256)] = acc[r];
    }
  }
}

// ---------------------------------------------------------------------------
// K2: main fused kernel. Per 16-edge tile (same batch b, same sender i,
// receivers j0..j0+15): build ez = gelu(P[i]+Q[j]+b_n2e) f16 tile in LDS,
// h = relu(ez @ W_e2v1 + b1) via 64x v_wmma_f32_16x16x32_f16,
// e2v = relu(h . w2 + b2) via fused shuffle reduction.
// 512 blocks x 4 waves, 4 tiles per wave (8192 tiles total).
// ---------------------------------------------------------------------------
__global__ void __launch_bounds__(128) k_edges(const float* __restrict__ P,
                                               const float* __restrict__ Q,
                                               const float* __restrict__ bn2e,
                                               const uint32_t* __restrict__ w1f,
                                               const float* __restrict__ b1,
                                               const float* __restrict__ w2,
                                               const float* __restrict__ b2p,
                                               float* __restrict__ e2v) {
  __shared__ __align__(16) uint16_t ez[4][16 * 256];
  int tid = threadIdx.x, lane = tid & 31, wave = tid >> 5;
  int half = lane >> 4, M = lane & 15;
  int waveGlobal = blockIdx.x * 4 + wave;
  float b2 = b2p[0];
  float b1v[8], w2v[8];
#pragma unroll
  for (int nt = 0; nt < 8; ++nt) {
    b1v[nt] = b1[nt * 16 + M];
    w2v[nt] = w2[nt * 16 + M];
  }
  float bv[8];
  ld8(bn2e, lane, bv);
  uint16_t* myez = ez[wave];
  const uint4* lx = (const uint4*)myez;
  const uint4* wf = (const uint4*)w1f;

  for (int it = 0; it < 4; ++it) {
    int tile = waveGlobal * 4 + it;
    int g0 = tile * 16;
    int b  = g0 >> 14;
    int e0 = g0 & 16383;
    int i  = e0 >> 7;
    int j0 = e0 & 127;
    if (it < 3) {  // global_prefetch_b8 of next tile's sender row
      int gn = (tile + 1) * 16;
      __builtin_prefetch(P + ((gn >> 14) * 128 + ((gn & 16383) >> 7)) * 256, 0, 0);
    }
    float pv[8];
    ld8(P + (b * 128 + i) * 256, lane, pv);
    for (int r = 0; r < 16; ++r) {
      float qv[8];
      ld8(Q + (b * 128 + j0 + r) * 256, lane, qv);
      uint4 o;
      uint32_t od[4];
#pragma unroll
      for (int t = 0; t < 4; ++t) {
        float a0 = gelu_exact(pv[2 * t]     + qv[2 * t]     + bv[2 * t]);
        float a1 = gelu_exact(pv[2 * t + 1] + qv[2 * t + 1] + bv[2 * t + 1]);
        od[t] = pack2h(a0, a1);
      }
      o.x = od[0]; o.y = od[1]; o.z = od[2]; o.w = od[3];
      ((uint4*)(myez + r * 256))[lane] = o;
    }
    __syncthreads();
    Frag16 afrag[8];
#pragma unroll
    for (int kt = 0; kt < 8; ++kt) {
      int el = M * 256 + kt * 32 + half * 8;
      afrag[kt].q[0] = lx[el >> 3];
      afrag[kt].q[1] = lx[(el + 16) >> 3];
    }
    float part[8] = {0.f, 0.f, 0.f, 0.f, 0.f, 0.f, 0.f, 0.f};
#pragma unroll
    for (int nt = 0; nt < 8; ++nt) {
      v8f acc = {0.f, 0.f, 0.f, 0.f, 0.f, 0.f, 0.f, 0.f};
#pragma unroll
      for (int kt = 0; kt < 8; ++kt) {
        Frag16 bf;
        int fi = ((kt * 8 + nt) * 32 + lane) * 2;
        bf.q[0] = wf[fi];
        bf.q[1] = wf[fi + 1];
        acc = __builtin_amdgcn_wmma_f32_16x16x32_f16(false, afrag[kt].h,
                                                     false, bf.h,
                                                     (short)0, acc, false, false);
      }
#pragma unroll
      for (int r = 0; r < 8; ++r) {
        float h = acc[r] + b1v[nt];
        h = h > 0.f ? h : 0.f;
        part[r] += h * w2v[nt];
      }
    }
#pragma unroll
    for (int r = 0; r < 8; ++r) {
      float s = part[r];
      s += __shfl_xor(s, 1);
      s += __shfl_xor(s, 2);
      s += __shfl_xor(s, 4);
      s += __shfl_xor(s, 8);
      part[r] = s;
    }
    if (M == 0) {
#pragma unroll
      for (int r = 0; r < 8; ++r) {
        float v = part[r] + b2;
        e2v[g0 + half * 8 + r] = v > 0.f ? v : 0.f;
      }
    }
    __syncthreads();
  }
}

// ---------------------------------------------------------------------------
// K3: per-row top-3 of r = (e < ALPHA ? BIG : e) over N=128 cols, with
// top_k-stable tie-break (lower index wins on equal values).
// One wave per row; butterfly shuffle merge.
// ---------------------------------------------------------------------------
__device__ __forceinline__ void ins3(float* v, int* ix, float nv, int ni) {
  bool g0 = (nv > v[0]) || (nv == v[0] && ni < ix[0]);
  bool g1 = (nv > v[1]) || (nv == v[1] && ni < ix[1]);
  bool g2 = (nv > v[2]) || (nv == v[2] && ni < ix[2]);
  if (g0)      { v[2]=v[1]; ix[2]=ix[1]; v[1]=v[0]; ix[1]=ix[0]; v[0]=nv; ix[0]=ni; }
  else if (g1) { v[2]=v[1]; ix[2]=ix[1]; v[1]=nv; ix[1]=ni; }
  else if (g2) { v[2]=nv; ix[2]=ni; }
}

__global__ void __launch_bounds__(128) k_top3(const float* __restrict__ e2v,
                                              float* __restrict__ tv,
                                              int* __restrict__ ti) {
  int tid = threadIdx.x, lane = tid & 31, wave = tid >> 5;
  int row = blockIdx.x * 4 + wave;                 // 0..1023
  const float* rp = e2v + row * 128;
  float v[3] = {-1e30f, -1e30f, -1e30f};
  int ix[3] = {1 << 30, 1 << 30, 1 << 30};
#pragma unroll
  for (int t = 0; t < 4; ++t) {
    int c = lane * 4 + t;
    float e = rp[c];
    ins3(v, ix, (e < ALPHA) ? BIG : e, c);
  }
#pragma unroll
  for (int m = 16; m >= 1; m >>= 1) {
#pragma unroll
    for (int t = 0; t < 3; ++t) {
      float ov = __shfl_xor(v[t], m);
      int   oi = __shfl_xor(ix[t], m);
      ins3(v, ix, ov, oi);
    }
  }
  if (lane == 0) {
#pragma unroll
    for (int t = 0; t < 3; ++t) {
      tv[row * 3 + t] = v[t];
      ti[row * 3 + t] = ix[t];
    }
  }
}

// ---------------------------------------------------------------------------
// K4: memory-bound epilogue. One wave per edge; recompute ez from P/Q,
// curvature k from top-3 tables, out = ez + 0.5*((k - exp(sigmoid(ez)))*ein).
// Streaming traffic (ein read, out write) uses non-temporal hints so the
// reused P/Q/top3 tables stay L2-resident.
// ---------------------------------------------------------------------------
__global__ void __launch_bounds__(256) k_out(const float* __restrict__ P,
                                             const float* __restrict__ Q,
                                             const float* __restrict__ bn2e,
                                             const float* __restrict__ ein,
                                             const float* __restrict__ e2v,
                                             const float* __restrict__ tv,
                                             const int* __restrict__ ti,
                                             float* __restrict__ out) {
  int tid = blockIdx.x * 256 + threadIdx.x;
  int g = tid >> 5;
  int lane = tid & 31;
  int b = g >> 14, e = g & 16383, i = e >> 7, j = e & 127;

  float ev = e2v[g];
  float k = 2.0f;
  if (ev >= ALPHA && ev < 1.0f) {
    int ri = (b * 128 + i) * 3, rj = (b * 128 + j) * 3;
    float mi = -INFINITY, mj = -INFINITY;
#pragma unroll
    for (int t = 0; t < 3; ++t) {
      int id = ti[ri + t]; float vv = tv[ri + t];
      if (id != i && id != j) mi = fmaxf(mi, vv);
      id = ti[rj + t]; vv = tv[rj + t];
      if (id != i && id != j) mj = fmaxf(mj, vv);
    }
    float di = fmaxf(mi, BIG), dj = fmaxf(mj, BIG);
    k = 2.0f - (sqrtf(ev / di) + sqrtf(ev / dj));
  }

  float pv[8], qv[8], bv[8], iv[8], ov[8];
  ld8(P + (b * 128 + i) * 256, lane, pv);
  ld8(Q + (b * 128 + j) * 256, lane, qv);
  ld8(bn2e, lane, bv);
  ld8_nt(ein + g * 256, lane, iv);
#pragma unroll
  for (int t = 0; t < 8; ++t) {
    float z = gelu_exact(pv[t] + qv[t] + bv[t]);
    float sig = 1.0f / (1.0f + expf(-z));
    ov[t] = z + BETA * ((k - expf(sig)) * iv[t]);
  }
  st8_nt(out + g * 256, lane, ov);
}

// ---------------------------------------------------------------------------
extern "C" void kernel_launch(void* const* d_in, const int* in_sizes, int n_in,
                              void* d_out, int out_size, void* d_ws, size_t ws_size,
                              hipStream_t stream) {
  const float* X    = (const float*)d_in[0];   // node_inputs [1024,256]
  const float* EIN  = (const float*)d_in[1];   // edges_input [131072,256]
  const float* Wn2e = (const float*)d_in[2];   // [512,256]
  const float* bn2e = (const float*)d_in[3];   // [256]
  const float* W1   = (const float*)d_in[4];   // [256,128]
  const float* b1   = (const float*)d_in[5];   // [128]
  const float* w2   = (const float*)d_in[6];   // [128,1]
  const float* b2   = (const float*)d_in[7];   // [1]
  (void)in_sizes; (void)n_in; (void)out_size; (void)ws_size;

  float* out = (float*)d_out;
  float* e2v = out + (size_t)8 * 128 * 128 * 256;   // second output, 131072 floats

  char* ws = (char*)d_ws;
  float*    Pm  = (float*)ws;                                 // 1 MB
  float*    Qm  = (float*)(ws + (1u << 20));                  // 1 MB
  uint32_t* w1f = (uint32_t*)(ws + (2u << 20));               // 64 KB
  uint32_t* wcf = (uint32_t*)(ws + (2u << 20) + (64u << 10)); // 256 KB
  float*    tv  = (float*)(ws + (2u << 20) + (320u << 10));   // 12 KB
  int*      ti  = (int*)((char*)tv + 1024 * 3 * 4);           // 12 KB

  k_swizzle<<<320, 256, 0, stream>>>(Wn2e, W1, w1f, wcf);
  k_pq<<<64, 128, 0, stream>>>(X, wcf, Pm, Qm);
  k_edges<<<512, 128, 0, stream>>>(Pm, Qm, bn2e, w1f, b1, w2, b2, e2v);
  k_top3<<<256, 128, 0, stream>>>(e2v, tv, ti);
  k_out<<<16384, 256, 0, stream>>>(Pm, Qm, bn2e, EIN, e2v, tv, ti, out);
}
